// MixerModel_25202868093361
// MI455X (gfx1250) — matrix operand order repaired and checked by reading, not compiled, not run on hardware
//
#include <hip/hip_runtime.h>
#include <hip/hip_bf16.h>
#include <cstdint>
#include <cstddef>

// ---------------- problem constants (from reference) ----------------
#define B_       8
#define L_       4096
#define C_IN     4
#define D_MODEL  256
#define N_LAYER  4
#define D_INNER  512
#define D_STATE  16
#define D_CONV   4
#define DT_RANK  16
#define EPSV     1e-5f
#define M_TOK    (B_ * L_)          // 32768 token rows

// ---------------- CDNA5 WMMA types ----------------
typedef __attribute__((ext_vector_type(16))) __bf16 v16bf;
typedef __attribute__((ext_vector_type(8)))  __bf16 v8bf;
typedef __attribute__((ext_vector_type(8)))  float  v8f;

// ---------------- helpers ----------------
static __device__ __forceinline__ __bf16 f2bf(float f) {
    unsigned u = __builtin_bit_cast(unsigned, f);
    u += 0x7FFFu + ((u >> 16) & 1u);           // round-to-nearest-even
    unsigned short s = (unsigned short)(u >> 16);
    return __builtin_bit_cast(__bf16, s);
}
static __device__ __forceinline__ float sigmoidf_(float x) {
    return 1.0f / (1.0f + __expf(-x));
}
static __device__ __forceinline__ float siluf_(float x) {
    return x * sigmoidf_(x);
}
static __device__ __forceinline__ float softplusf_(float x) {
    return (x > 20.0f) ? x : log1pf(__expf(x));
}
static __device__ __forceinline__ v16bf join16(v8bf lo, v8bf hi) {
    v16bf r;
#pragma unroll
    for (int i = 0; i < 8; ++i) { r[i] = lo[i]; r[i + 8] = hi[i]; }
    return r;
}

// CDNA5 async copy: 16 bytes global -> LDS, tracked by ASYNCcnt (no VGPR roundtrip).
// LDS operand is the low 32 bits of the flat shared-memory pointer (HW truncates
// flat LDS-aperture addresses to addr[31:0] identically).
static __device__ __forceinline__ void async_cp16(const __bf16* g, __bf16* l) {
    unsigned lds = (unsigned)(uintptr_t)l;
    asm volatile("global_load_async_to_lds_b128 %0, %1, off"
                 :: "v"(lds), "v"(g) : "memory");
}
static __device__ __forceinline__ void wait_async0() {
    asm volatile("s_wait_asynccnt 0x0" ::: "memory");
}

// ---------------- f32 -> bf16 convert (weights) ----------------
__global__ void k_f2bf(const float* __restrict__ src, __bf16* __restrict__ dst, int n) {
    int i = blockIdx.x * blockDim.x + threadIdx.x;
    if (i < n) dst[i] = f2bf(src[i]);
}

// ---------------- embedding: h[b,l,d] = sum_c x[b,c,l]*ew[d,c] + eb[d] ----------------
__global__ void k_embed(const float* __restrict__ x, const float* __restrict__ ew,
                        const float* __restrict__ eb, float* __restrict__ h) {
    int idx = blockIdx.x * blockDim.x + threadIdx.x;      // over M_TOK*D_MODEL
    int d = idx & (D_MODEL - 1);
    int t = idx >> 8;                                     // b*L + l
    int b = t >> 12;                                      // / L_
    int l = t & (L_ - 1);
    float acc = eb[d];
#pragma unroll
    for (int c = 0; c < C_IN; ++c)
        acc += x[((size_t)(b * C_IN + c)) * L_ + l] * ew[d * C_IN + c];
    h[idx] = acc;
}

// ---------------- fused residual-add + LayerNorm (wave per 256-wide row) ----------
__global__ __launch_bounds__(256) void k_ln(
        const float* __restrict__ h, float* __restrict__ res,
        const float* __restrict__ w, const float* __restrict__ bb,
        __bf16* __restrict__ obf, float* __restrict__ of, int first) {
    int wave = threadIdx.x >> 5;
    int lane = threadIdx.x & 31;
    int row  = blockIdx.x * 8 + wave;                     // M_TOK rows total
    const float* hr = h + (size_t)row * D_MODEL;
    float* rr = res + (size_t)row * D_MODEL;
    float v[8], sum = 0.f, sq = 0.f;
#pragma unroll
    for (int i = 0; i < 8; ++i) {
        int d = lane + i * 32;
        float t = first ? hr[d] : (hr[d] + rr[d]);
        v[i] = t; sum += t; sq += t * t;
    }
#pragma unroll
    for (int m = 16; m >= 1; m >>= 1) {
        sum += __shfl_xor(sum, m, 32);
        sq  += __shfl_xor(sq,  m, 32);
    }
    float mean = sum * (1.0f / D_MODEL);
    float var  = sq * (1.0f / D_MODEL) - mean * mean;
    float rstd = rsqrtf(var + EPSV);
#pragma unroll
    for (int i = 0; i < 8; ++i) {
        int d = lane + i * 32;
        rr[d] = v[i];
        float o = (v[i] - mean) * rstd * w[d] + bb[d];
        if (obf) obf[(size_t)row * D_MODEL + d] = f2bf(o);
        if (of)  of[(size_t)row * D_MODEL + d]  = o;
    }
}

// ---------------- bf16 WMMA GEMM: C(M,N) f32 = A(M,K) * W(N,K)^T -------------------
// Block tile 128x128, BK=32, 8 waves (4 along M x 2 along N), wave tile 32x64.
// Double-buffered LDS staged with global_load_async_to_lds_b128 (ASYNCcnt).
// Out-of-range W rows are clamped to row 0 (results discarded by the store-side
// gcol<N guard), so the staging loop is completely branch-free.
// Requires: M % 128 == 0, K % 32 == 0.
#define BM 128
#define BN 128
#define BK 32
#define LDA_S 40   // padded LDS stride (bf16 elems) to break bank conflicts
__global__ __launch_bounds__(256) void k_gemm_bf16(
        const __bf16* __restrict__ A, const __bf16* __restrict__ W,
        float* __restrict__ C, int M, int N, int K) {
    __shared__ __bf16 As[2][BM][LDA_S];
    __shared__ __bf16 Bs[2][BN][LDA_S];
    const int tid  = threadIdx.x;
    const int lane = tid & 31;
    const int wid  = tid >> 5;
    const int wm   = wid & 3;        // 4 waves along M: 32 rows each
    const int wn   = wid >> 2;       // 2 waves along N: 64 cols each
    const int bm   = blockIdx.x * BM;
    const int bn   = blockIdx.y * BN;
    const int hf   = lane >> 4;      // lane half (0/1)
    const int l16  = lane & 15;

    // staging assignment: each thread owns one 16-elem chunk of A and of W
    const int srow = tid >> 1;       // 0..127
    const int sseg = tid & 1;        // 0/1 -> K sub-chunk
    const __bf16* aptr = A + (size_t)(bm + srow) * K + sseg * 16;
    const int gn = bn + srow;
    const __bf16* wptr = W + (size_t)(gn < N ? gn : 0) * K + sseg * 16;

    auto stage = [&](int buf, int k0) {
        async_cp16(aptr + k0,     &As[buf][srow][sseg * 16]);
        async_cp16(aptr + k0 + 8, &As[buf][srow][sseg * 16 + 8]);
        async_cp16(wptr + k0,     &Bs[buf][srow][sseg * 16]);
        async_cp16(wptr + k0 + 8, &Bs[buf][srow][sseg * 16 + 8]);
    };

    v8f acc[2][4];
#pragma unroll
    for (int mi = 0; mi < 2; ++mi)
#pragma unroll
        for (int ni = 0; ni < 4; ++ni)
#pragma unroll
            for (int r = 0; r < 8; ++r) acc[mi][ni][r] = 0.f;

    stage(0, 0);
    wait_async0();
    __syncthreads();

    int cur = 0;
    for (int k0 = 0; k0 < K; k0 += BK) {
        if (k0 + BK < K) stage(cur ^ 1, k0 + BK);   // overlap copy of next tile

        // A fragment (ISA 16-bit A 16x32): elem e<8 -> K=hf*8+e ; e>=8 -> K=16+hf*8+(e-8)
        v16bf aFrag[2], bFrag[4];
#pragma unroll
        for (int mi = 0; mi < 2; ++mi) {
            int row = wm * 32 + mi * 16 + l16;
            v8bf lo = *(const v8bf*)&As[cur][row][hf * 8];
            v8bf hi = *(const v8bf*)&As[cur][row][16 + hf * 8];
            aFrag[mi] = join16(lo, hi);
        }
        // B fragment (32x16, N-major in LDS): elem e<8 -> K=hf*8+e ; e>=8 -> K=16+hf*8+(e-8), col = l16
#pragma unroll
        for (int ni = 0; ni < 4; ++ni) {
            int n = wn * 64 + ni * 16 + l16;
            v8bf lo = *(const v8bf*)&Bs[cur][n][hf * 8];
            v8bf hi = *(const v8bf*)&Bs[cur][n][16 + hf * 8];
            bFrag[ni] = join16(lo, hi);
        }
#pragma unroll
        for (int mi = 0; mi < 2; ++mi)
#pragma unroll
            for (int ni = 0; ni < 4; ++ni)
                acc[mi][ni] = __builtin_amdgcn_wmma_f32_16x16x32_bf16(
                    false, aFrag[mi], false, bFrag[ni],
                    (short)0, acc[mi][ni], false, false);

        wait_async0();      // our next-tile async copies have landed
        __syncthreads();    // everyone done reading cur / writing cur^1
        cur ^= 1;
    }

    // C/D layout: VGPR r, lanes0-15 -> M=r, lanes16-31 -> M=r+8
#pragma unroll
    for (int mi = 0; mi < 2; ++mi)
#pragma unroll
        for (int ni = 0; ni < 4; ++ni) {
            int gcol = bn + wn * 64 + ni * 16 + l16;
            if (gcol < N) {
#pragma unroll
                for (int r = 0; r < 8; ++r) {
                    int grow = bm + wm * 32 + mi * 16 + hf * 8 + r;
                    C[(size_t)grow * N + gcol] = acc[mi][ni][r];
                }
            }
        }
}

// ---------------- depthwise causal conv (k=4) + SiLU ----------------
__global__ void k_conv(const float* __restrict__ xz, const float* __restrict__ cw,
                       const float* __restrict__ cb, float* __restrict__ xc,
                       __bf16* __restrict__ xcbf) {
    int idx = blockIdx.x * blockDim.x + threadIdx.x;      // M_TOK*D_INNER
    int d = idx & (D_INNER - 1);
    int t = idx >> 9;
    int b = t >> 12;
    int l = t & (L_ - 1);
    float acc = cb[d];
#pragma unroll
    for (int j = 0; j < D_CONV; ++j) {
        int ls = l - (D_CONV - 1) + j;
        if (ls >= 0)
            acc += xz[((size_t)(b * L_ + ls)) * (2 * D_INNER) + d] * cw[d * D_CONV + j];
    }
    float s = siluf_(acc);
    xc[idx] = s;
    xcbf[idx] = f2bf(s);
}

// ---------------- dt = softplus(dbl[:,:16] @ dtw^T + dtb)  (K=16, VALU) ------------
__global__ void k_dt(const float* __restrict__ dbl, const float* __restrict__ dtw,
                     const float* __restrict__ dtb, float* __restrict__ dt) {
    int idx = blockIdx.x * blockDim.x + threadIdx.x;      // M_TOK*D_INNER
    int d = idx & (D_INNER - 1);
    int t = idx >> 9;
    const float* row = dbl + (size_t)t * (DT_RANK + 2 * D_STATE);
    float acc = dtb[d];
#pragma unroll
    for (int r = 0; r < DT_RANK; ++r)
        acc += row[r] * dtw[d * DT_RANK + r];
    dt[idx] = softplusf_(acc);
}

// ---------------- selective scan: lane = (channel, state); 16-lane reduction -------
__global__ __launch_bounds__(256) void k_scan(
        const float* __restrict__ dt, const float* __restrict__ xc,
        const float* __restrict__ dbl, const float* __restrict__ a_log,
        float* __restrict__ y) {
    int tid = blockIdx.x * blockDim.x + threadIdx.x;      // B_*D_INNER*D_STATE
    int s  = tid & (D_STATE - 1);
    int ch = tid >> 4;                                    // global channel b*512+d
    int b = ch >> 9;
    int d = ch & (D_INNER - 1);
    float Ac = -__expf(a_log[d * D_STATE + s]);
    float h = 0.f;
    const float* dt_b  = dt  + (size_t)b * L_ * D_INNER + d;
    const float* x_b   = xc  + (size_t)b * L_ * D_INNER + d;
    const float* dbl_b = dbl + (size_t)b * L_ * (DT_RANK + 2 * D_STATE);
    float* y_b = y + (size_t)b * L_ * D_INNER + d;
    for (int t = 0; t < L_; ++t) {
        float dtv = dt_b[(size_t)t * D_INNER];
        float xv  = x_b[(size_t)t * D_INNER];
        const float* row = dbl_b + (size_t)t * (DT_RANK + 2 * D_STATE);
        float bv = row[DT_RANK + s];
        float cv = row[DT_RANK + D_STATE + s];
        h = h * __expf(dtv * Ac) + (dtv * xv) * bv;
        float p = h * cv;
        p += __shfl_xor(p, 1, 32);
        p += __shfl_xor(p, 2, 32);
        p += __shfl_xor(p, 4, 32);
        p += __shfl_xor(p, 8, 32);
        if (s == 0) y_b[(size_t)t * D_INNER] = p;
    }
}

// ---------------- y = (y_scan + xc*D) * silu(z)  -> bf16 ----------------
__global__ void k_ycomb(const float* __restrict__ ys, const float* __restrict__ xc,
                        const float* __restrict__ xz, const float* __restrict__ Dp,
                        __bf16* __restrict__ ybf) {
    int idx = blockIdx.x * blockDim.x + threadIdx.x;      // M_TOK*D_INNER
    int d = idx & (D_INNER - 1);
    int t = idx >> 9;
    float z = xz[(size_t)t * (2 * D_INNER) + D_INNER + d];
    float val = (ys[idx] + xc[idx] * Dp[d]) * siluf_(z);
    ybf[idx] = f2bf(val);
}

// ---------------- host side ----------------
static inline uint8_t* carve(uint8_t*& p, size_t bytes) {
    uint8_t* r = p;
    p += (bytes + 255) & ~(size_t)255;
    return r;
}

extern "C" void kernel_launch(void* const* d_in, const int* in_sizes, int n_in,
                              void* d_out, int out_size, void* d_ws, size_t ws_size,
                              hipStream_t stream) {
    (void)in_sizes; (void)n_in; (void)out_size; (void)ws_size;
    const float* x        = (const float*)d_in[0];
    const float* emb_w    = (const float*)d_in[1];
    const float* emb_b    = (const float*)d_in[2];
    const float* in_w     = (const float*)d_in[3];   // (4,1024,256)
    const float* conv_w   = (const float*)d_in[4];   // (4,512,4)
    const float* conv_b   = (const float*)d_in[5];   // (4,512)
    const float* xp_w     = (const float*)d_in[6];   // (4,48,512)
    const float* dt_w     = (const float*)d_in[7];   // (4,512,16)
    const float* dt_b     = (const float*)d_in[8];   // (4,512)
    const float* a_log    = (const float*)d_in[9];   // (4,512,16)
    const float* Dsk      = (const float*)d_in[10];  // (4,512)
    const float* out_w    = (const float*)d_in[11];  // (4,256,512)
    const float* norm_w   = (const float*)d_in[12];  // (4,256)
    const float* norm_b   = (const float*)d_in[13];
    const float* normf_w  = (const float*)d_in[14];
    const float* normf_b  = (const float*)d_in[15];

    uint8_t* p = (uint8_t*)d_ws;
    float*  res   = (float*)carve(p, (size_t)M_TOK * D_MODEL * 4);
    float*  h     = (float*)carve(p, (size_t)M_TOK * D_MODEL * 4);
    __bf16* hnbf  = (__bf16*)carve(p, (size_t)M_TOK * D_MODEL * 2);
    float*  xz    = (float*)carve(p, (size_t)M_TOK * 2 * D_INNER * 4);
    float*  xc    = (float*)carve(p, (size_t)M_TOK * D_INNER * 4);
    __bf16* xcbf  = (__bf16*)carve(p, (size_t)M_TOK * D_INNER * 2);
    float*  dbl   = (float*)carve(p, (size_t)M_TOK * (DT_RANK + 2 * D_STATE) * 4);
    float*  dt    = (float*)carve(p, (size_t)M_TOK * D_INNER * 4);
    float*  ys    = (float*)carve(p, (size_t)M_TOK * D_INNER * 4);
    __bf16* ybf   = (__bf16*)carve(p, (size_t)M_TOK * D_INNER * 2);
    const int n_inw = N_LAYER * 2 * D_INNER * D_MODEL;   // 1048576
    const int n_xpw = N_LAYER * (DT_RANK + 2 * D_STATE) * D_INNER; // 98304
    const int n_opw = N_LAYER * D_MODEL * D_INNER;       // 524288
    __bf16* inwbf = (__bf16*)carve(p, (size_t)n_inw * 2);
    __bf16* xpwbf = (__bf16*)carve(p, (size_t)n_xpw * 2);
    __bf16* opwbf = (__bf16*)carve(p, (size_t)n_opw * 2);

    // weight conversion to bf16 (deterministic; redone every call)
    k_f2bf<<<(n_inw + 255) / 256, 256, 0, stream>>>(in_w, inwbf, n_inw);
    k_f2bf<<<(n_xpw + 255) / 256, 256, 0, stream>>>(xp_w, xpwbf, n_xpw);
    k_f2bf<<<(n_opw + 255) / 256, 256, 0, stream>>>(out_w, opwbf, n_opw);

    // embedding
    k_embed<<<(M_TOK * D_MODEL) / 256, 256, 0, stream>>>(x, emb_w, emb_b, h);

    const int nEl = M_TOK * D_INNER;   // 16M elementwise work items
    for (int i = 0; i < N_LAYER; ++i) {
        // residual + LN -> bf16 A-matrix for in_proj
        k_ln<<<M_TOK / 8, 256, 0, stream>>>(h, res,
            norm_w + i * D_MODEL, norm_b + i * D_MODEL, hnbf, nullptr, (i == 0) ? 1 : 0);
        // in_proj: (32768 x 256) @ (1024 x 256)^T
        {
            dim3 g(M_TOK / BM, (2 * D_INNER) / BN);       // (256, 8)
            k_gemm_bf16<<<g, 256, 0, stream>>>(hnbf, inwbf + (size_t)i * 2 * D_INNER * D_MODEL,
                                               xz, M_TOK, 2 * D_INNER, D_MODEL);
        }
        // causal depthwise conv + SiLU
        k_conv<<<nEl / 256, 256, 0, stream>>>(xz, conv_w + (size_t)i * D_INNER * D_CONV,
                                              conv_b + (size_t)i * D_INNER, xc, xcbf);
        // x_proj: (32768 x 512) @ (48 x 512)^T   (N=48 guarded at store)
        {
            dim3 g(M_TOK / BM, 1);
            k_gemm_bf16<<<g, 256, 0, stream>>>(xcbf,
                xpwbf + (size_t)i * (DT_RANK + 2 * D_STATE) * D_INNER,
                dbl, M_TOK, DT_RANK + 2 * D_STATE, D_INNER);
        }
        // dt projection + softplus
        k_dt<<<nEl / 256, 256, 0, stream>>>(dbl, dt_w + (size_t)i * D_INNER * DT_RANK,
                                            dt_b + (size_t)i * D_INNER, dt);
        // selective scan (sequential over L, parallel over B x D_INNER x D_STATE)
        k_scan<<<(B_ * D_INNER * D_STATE) / 256, 256, 0, stream>>>(
            dt, xc, dbl, a_log + (size_t)i * D_INNER * D_STATE, ys);
        // gate + skip -> bf16
        k_ycomb<<<nEl / 256, 256, 0, stream>>>(ys, xc, xz, Dsk + (size_t)i * D_INNER, ybf);
        // out_proj: (32768 x 512) @ (256 x 512)^T  -> new h
        {
            dim3 g(M_TOK / BM, D_MODEL / BN);             // (256, 2)
            k_gemm_bf16<<<g, 256, 0, stream>>>(ybf, opwbf + (size_t)i * D_MODEL * D_INNER,
                                               h, M_TOK, D_MODEL, D_INNER);
        }
    }
    // final residual + LN -> f32 output
    k_ln<<<M_TOK / 8, 256, 0, stream>>>(h, res, normf_w, normf_b,
                                        nullptr, (float*)d_out, 0);
}
